// TrajGRUcell_71914932404492
// MI455X (gfx1250) — matrix operand order repaired and checked by reading, not compiled, not run on hardware
//
#include <hip/hip_runtime.h>

typedef __bf16  v16bf  __attribute__((ext_vector_type(16)));
typedef __bf16  bf16x8 __attribute__((ext_vector_type(8)));
typedef float   v8f    __attribute__((ext_vector_type(8)));

typedef __attribute__((address_space(3))) char       lds_char;
typedef __attribute__((address_space(3))) const v16bf lds_v16bf;

#define HW     16384      // 128*128
#define WID    128
#define BATCH  4
#define CIN    8
#define COUT   64
#define LVL    13
#define KDIM   832        // LVL * COUT
#define BROW   1680       // LDS bytes per staged B pixel-row (1664 data + 16 pad)

__device__ __forceinline__ float leaky2(float t) { return t >= 0.0f ? t : 0.2f * t; }
__device__ __forceinline__ float sigmf(float t)  { return 1.0f / (1.0f + __expf(-t)); }

// ---------------------------------------------------------------------------
// Kernel 1: 5x5 conv, 72 stacked channels (x:8 + prev:64) -> 32, pad 2, leaky
// ---------------------------------------------------------------------------
__global__ __launch_bounds__(256)
void conv1_kernel(const float* __restrict__ x, const float* __restrict__ prev,
                  const float* __restrict__ w1, const float* __restrict__ b1,
                  float* __restrict__ h1)
{
    __shared__ float wsm[72 * 25];
    const int co  = blockIdx.y;            // 0..31
    const int b   = blockIdx.z;            // 0..3
    const int pix = blockIdx.x * 256 + threadIdx.x;
    for (int i = threadIdx.x; i < 72 * 25; i += 256)
        wsm[i] = w1[co * 72 * 25 + i];
    __syncthreads();

    const int h = pix >> 7, w = pix & 127;
    float acc = b1[co];
    for (int cin = 0; cin < 72; ++cin) {
        const float* src = (cin < CIN)
            ? (x    + ((size_t)b * CIN  + cin      ) * HW)
            : (prev + ((size_t)b * COUT + (cin - 8)) * HW);
        const float* wp = &wsm[cin * 25];
        #pragma unroll
        for (int dy = 0; dy < 5; ++dy) {
            const int hh = h + dy - 2;
            if (hh < 0 || hh >= WID) continue;
            #pragma unroll
            for (int dx = 0; dx < 5; ++dx) {
                const int ww = w + dx - 2;
                if (ww < 0 || ww >= WID) continue;
                acc = fmaf(src[hh * WID + ww], wp[dy * 5 + dx], acc);
            }
        }
    }
    h1[((size_t)b * 32 + co) * HW + pix] = leaky2(acc);
}

// ---------------------------------------------------------------------------
// Kernel 2: 5x5 conv, 32 -> 26 (flow), pad 2
// ---------------------------------------------------------------------------
__global__ __launch_bounds__(256)
void conv2_kernel(const float* __restrict__ h1,
                  const float* __restrict__ w2, const float* __restrict__ b2,
                  float* __restrict__ flow)
{
    __shared__ float wsm[32 * 25];
    const int co  = blockIdx.y;            // 0..25
    const int b   = blockIdx.z;
    const int pix = blockIdx.x * 256 + threadIdx.x;
    for (int i = threadIdx.x; i < 32 * 25; i += 256)
        wsm[i] = w2[co * 32 * 25 + i];
    __syncthreads();

    const int h = pix >> 7, w = pix & 127;
    float acc = b2[co];
    for (int cin = 0; cin < 32; ++cin) {
        const float* src = h1 + ((size_t)b * 32 + cin) * HW;
        const float* wp  = &wsm[cin * 25];
        #pragma unroll
        for (int dy = 0; dy < 5; ++dy) {
            const int hh = h + dy - 2;
            if (hh < 0 || hh >= WID) continue;
            #pragma unroll
            for (int dx = 0; dx < 5; ++dx) {
                const int ww = w + dx - 2;
                if (ww < 0 || ww >= WID) continue;
                acc = fmaf(src[hh * WID + ww], wp[dy * 5 + dx], acc);
            }
        }
    }
    flow[((size_t)b * 26 + co) * HW + pix] = acc;
}

// ---------------------------------------------------------------------------
// Kernel 3: repack {rgw,ugw,ogw} f32 [64][832] -> bf16 Wc[192][832] + biases
// ---------------------------------------------------------------------------
__global__ __launch_bounds__(256)
void wpack_kernel(const float* __restrict__ rgw, const float* __restrict__ ugw,
                  const float* __restrict__ ogw,
                  const float* __restrict__ rgb, const float* __restrict__ ugb,
                  const float* __restrict__ ogb,
                  __bf16* __restrict__ Wc, float* __restrict__ wcb)
{
    const int idx = blockIdx.x * 256 + threadIdx.x;
    if (idx < 192 * KDIM) {
        const int row = idx / KDIM, k = idx - row * KDIM;
        const int g = row >> 6, c = row & 63;
        const float* src = (g == 0) ? rgw : (g == 1) ? ugw : ogw;
        Wc[idx] = (__bf16)src[c * KDIM + k];
    }
    if (idx < 192) {
        const int g = idx >> 6, c = idx & 63;
        wcb[idx] = (g == 0) ? rgb[c] : (g == 1) ? ugb[c] : ogb[c];
    }
}

// ---------------------------------------------------------------------------
// Kernel 4: fused 3x3 convs of x with {rg,ug,og} -> xg[B][192][HW] (bias in)
// ---------------------------------------------------------------------------
__global__ __launch_bounds__(256)
void xconv_kernel(const float* __restrict__ x,
                  const float* __restrict__ rgw, const float* __restrict__ rgb,
                  const float* __restrict__ ugw, const float* __restrict__ ugb,
                  const float* __restrict__ ogw, const float* __restrict__ ogb,
                  float* __restrict__ xg)
{
    __shared__ float wsm[CIN * 9];
    const int gc  = blockIdx.y;            // 0..191
    const int b   = blockIdx.z;
    const int pix = blockIdx.x * 256 + threadIdx.x;
    const int g = gc >> 6, c = gc & 63;
    const float* wsrc = (g == 0) ? rgw : (g == 1) ? ugw : ogw;
    const float* bsrc = (g == 0) ? rgb : (g == 1) ? ugb : ogb;
    if (threadIdx.x < CIN * 9) wsm[threadIdx.x] = wsrc[c * CIN * 9 + threadIdx.x];
    __syncthreads();

    const int h = pix >> 7, w = pix & 127;
    float acc = bsrc[c];
    for (int cin = 0; cin < CIN; ++cin) {
        const float* src = x + ((size_t)b * CIN + cin) * HW;
        const float* wp  = &wsm[cin * 9];
        #pragma unroll
        for (int dy = 0; dy < 3; ++dy) {
            const int hh = h + dy - 1;
            if (hh < 0 || hh >= WID) continue;
            #pragma unroll
            for (int dx = 0; dx < 3; ++dx) {
                const int ww = w + dx - 1;
                if (ww < 0 || ww >= WID) continue;
                acc = fmaf(src[hh * WID + ww], wp[dy * 3 + dx], acc);
            }
        }
    }
    xg[((size_t)b * 192 + gc) * HW + pix] = acc;
}

// ---------------------------------------------------------------------------
// Kernel 5: grid sample of prev_state (zero pad, bilinear), L warps,
//           writes M pixel-major: M[b][pix][l*64 + c] as bf16
// ---------------------------------------------------------------------------
__global__ __launch_bounds__(256)
void sample_kernel(const float* __restrict__ flow, const float* __restrict__ prev,
                   __bf16* __restrict__ M)
{
    const int pix = blockIdx.x * 256 + threadIdx.x;
    const int l   = blockIdx.y;            // 0..12
    const int b   = blockIdx.z;
    const int h = pix >> 7, w = pix & 127;

    const float u = flow[((size_t)b * 26 + l      ) * HW + pix];
    const float v = flow[((size_t)b * 26 + LVL + l) * HW + pix];
    const float fx = (float)w + (u * (float)WID - 1.0f) * 0.5f;
    const float fy = (float)h + (v * (float)WID - 1.0f) * 0.5f;
    const float x0f = floorf(fx), y0f = floorf(fy);
    const float wx1 = fx - x0f, wy1 = fy - y0f;
    const float wx0 = 1.0f - wx1, wy0 = 1.0f - wy1;
    const int x0 = (int)x0f, y0 = (int)y0f, x1 = x0 + 1, y1 = y0 + 1;

    const float mx0 = (x0 >= 0 && x0 < WID) ? 1.0f : 0.0f;
    const float mx1 = (x1 >= 0 && x1 < WID) ? 1.0f : 0.0f;
    const float my0 = (y0 >= 0 && y0 < WID) ? 1.0f : 0.0f;
    const float my1 = (y1 >= 0 && y1 < WID) ? 1.0f : 0.0f;
    const int x0c = min(max(x0, 0), WID - 1), x1c = min(max(x1, 0), WID - 1);
    const int y0c = min(max(y0, 0), WID - 1), y1c = min(max(y1, 0), WID - 1);
    const int o00 = y0c * WID + x0c, o10 = y0c * WID + x1c;
    const int o01 = y1c * WID + x0c, o11 = y1c * WID + x1c;
    const float W00 = wx0 * wy0 * mx0 * my0, W10 = wx1 * wy0 * mx1 * my0;
    const float W01 = wx0 * wy1 * mx0 * my1, W11 = wx1 * wy1 * mx1 * my1;

    const float* base = prev + (size_t)b * COUT * HW;
    __bf16* dst = M + ((size_t)b * HW + pix) * KDIM + l * COUT;
    #pragma unroll 4
    for (int c = 0; c < COUT; ++c) {
        const float* p = base + (size_t)c * HW;
        const float val = p[o00] * W00 + p[o10] * W10 + p[o01] * W01 + p[o11] * W11;
        dst[c] = (__bf16)val;
    }
}

// ---------------------------------------------------------------------------
// Kernel 6: WMMA GEMM (Wc[192x832] x M[832 x N]) + fused GRU epilogue.
// B tile (16 pixels x 832 K, bf16) is staged once per block into LDS via the
// gfx1250 async-to-LDS datapath (ASYNCcnt), then all 4 waves feed their WMMA
// B fragments from ds_load_b128. A (weights) streams from global/L2.
// ---------------------------------------------------------------------------
__global__ __launch_bounds__(128)
void gemm_gru_kernel(const __bf16* __restrict__ M,   // [B][HW][K]
                     const __bf16* __restrict__ Wc,  // [192][K]
                     const float*  __restrict__ wcb, // [192]
                     const float*  __restrict__ xg,  // [B][192][HW]
                     const float*  __restrict__ prev,// [B][64][HW]
                     float*        __restrict__ out) // [B][64][HW]
{
    __shared__ alignas(32) char Btile[16 * BROW];    // 26,880 B

    const int tid  = threadIdx.x;
    const int lane = tid & 31;
    const int wave = tid >> 5;             // 4 channel tiles per block
    const int p0   = blockIdx.x * 16;      // pixel tile base
    const int b    = blockIdx.y;
    const int c0   = wave * 16;            // channel tile base (0..48)

    const int half = lane >> 4;            // 0/1 -> K-half
    const int rlo  = lane & 15;            // A-row / B-column within tile

    // --- Stage B tile: 16 rows x 1664 B, async global->LDS, 16 B per transfer.
    // thread t copies row (t>>3), byte chunks ((t&7) + 8j)*16, j = 0..12.
    lds_char* lbase = (lds_char*)Btile;
    {
        const int srow = tid >> 3;                   // 0..15
        const char* grow = (const char*)(M + ((size_t)b * HW + p0 + srow) * KDIM);
        const unsigned lrow = (unsigned)(unsigned long long)(lbase + srow * BROW);
        #pragma unroll
        for (int j = 0; j < 13; ++j) {
            const int off = ((tid & 7) + 8 * j) * 16;
            const unsigned laddr = lrow + off;
            const char* gaddr = grow + off;
            asm volatile("global_load_async_to_lds_b128 %0, %1, off"
                         :: "v"(laddr), "v"(gaddr) : "memory");
        }
    }
    asm volatile("s_wait_asynccnt 0x0" ::: "memory");
    __syncthreads();

    v8f acc0 = {}, acc1 = {}, acc2 = {};

    // B fragment: lane holds column (pixel p0+rlo), K = 32t + 16*half .. +15
    lds_char* Bl = lbase + rlo * BROW + half * 32;
    // A fragments: lane holds row, K = 32t + 16*half .. +15
    const __bf16* A0 = Wc + (size_t)(  0 + c0 + rlo) * KDIM + half * 16;
    const __bf16* A1 = Wc + (size_t)( 64 + c0 + rlo) * KDIM + half * 16;
    const __bf16* A2 = Wc + (size_t)(128 + c0 + rlo) * KDIM + half * 16;

    for (int t = 0; t < KDIM / 32; ++t) {
        const v16bf bfrag = *(lds_v16bf*)(Bl + t * 64);

        bf16x8 lo = *(const bf16x8*)(A0 + t * 32);
        bf16x8 hi = *(const bf16x8*)(A0 + t * 32 + 8);
        v16bf a = __builtin_shufflevector(lo, hi, 0,1,2,3,4,5,6,7,8,9,10,11,12,13,14,15);
        acc0 = __builtin_amdgcn_wmma_f32_16x16x32_bf16(false, a, false, bfrag,
                                                       (short)0, acc0, false, false);
        lo = *(const bf16x8*)(A1 + t * 32);
        hi = *(const bf16x8*)(A1 + t * 32 + 8);
        a  = __builtin_shufflevector(lo, hi, 0,1,2,3,4,5,6,7,8,9,10,11,12,13,14,15);
        acc1 = __builtin_amdgcn_wmma_f32_16x16x32_bf16(false, a, false, bfrag,
                                                       (short)0, acc1, false, false);
        lo = *(const bf16x8*)(A2 + t * 32);
        hi = *(const bf16x8*)(A2 + t * 32 + 8);
        a  = __builtin_shufflevector(lo, hi, 0,1,2,3,4,5,6,7,8,9,10,11,12,13,14,15);
        acc2 = __builtin_amdgcn_wmma_f32_16x16x32_bf16(false, a, false, bfrag,
                                                       (short)0, acc2, false, false);
    }

    // Epilogue. C layout: lanes 0-15 col N=lane, row = VGPR r; lanes 16-31 row += 8.
    const int pix = p0 + rlo;
    #pragma unroll
    for (int r = 0; r < 8; ++r) {
        const int c = c0 + r + half * 8;   // output channel 0..63
        const float xr = xg[((size_t)b * 192 +       c) * HW + pix];
        const float xu = xg[((size_t)b * 192 +  64 + c) * HW + pix];
        const float xo = xg[((size_t)b * 192 + 128 + c) * HW + pix];
        const float reset  = sigmf(xr + acc0[r] + wcb[c]);
        const float update = sigmf(xu + acc1[r] + wcb[64 + c]);
        const float ogwv   = leaky2(acc2[r] + wcb[128 + c]);
        const float ogv    = leaky2(xo);
        const float oi     = leaky2(ogv + reset * ogwv);
        const float pv     = prev[((size_t)b * COUT + c) * HW + pix];
        out[((size_t)b * COUT + c) * HW + pix] = oi * (1.0f - update) + update * pv;
    }
}

// ---------------------------------------------------------------------------
extern "C" void kernel_launch(void* const* d_in, const int* in_sizes, int n_in,
                              void* d_out, int out_size, void* d_ws, size_t ws_size,
                              hipStream_t stream) {
    const float* x       = (const float*)d_in[0];
    const float* prev    = (const float*)d_in[1];
    const float* disp_w1 = (const float*)d_in[2];
    const float* disp_b1 = (const float*)d_in[3];
    const float* disp_w2 = (const float*)d_in[4];
    const float* disp_b2 = (const float*)d_in[5];
    const float* rg_w    = (const float*)d_in[6];
    const float* rg_b    = (const float*)d_in[7];
    const float* ug_w    = (const float*)d_in[8];
    const float* ug_b    = (const float*)d_in[9];
    const float* og_w    = (const float*)d_in[10];
    const float* og_b    = (const float*)d_in[11];
    const float* rgw_w   = (const float*)d_in[12];
    const float* rgw_b   = (const float*)d_in[13];
    const float* ugw_w   = (const float*)d_in[14];
    const float* ugw_b   = (const float*)d_in[15];
    const float* ogw_w   = (const float*)d_in[16];
    const float* ogw_b   = (const float*)d_in[17];
    float* out = (float*)d_out;

    char* ws = (char*)d_ws;
    float*  h1   = (float*)(ws);                         // 8,388,608 B
    float*  flow = (float*)(ws + 8388608);               // 6,815,744 B
    float*  xg   = (float*)(ws + 15204352);              // 50,331,648 B
    __bf16* M    = (__bf16*)(ws + 65536000);             // 109,051,904 B
    __bf16* Wc   = (__bf16*)(ws + 174587904);            // 319,488 B
    float*  wcb  = (float*)(ws + 174907392);             // 768 B

    conv1_kernel<<<dim3(HW / 256, 32, BATCH), 256, 0, stream>>>(
        x, prev, disp_w1, disp_b1, h1);
    conv2_kernel<<<dim3(HW / 256, 26, BATCH), 256, 0, stream>>>(
        h1, disp_w2, disp_b2, flow);
    wpack_kernel<<<dim3((192 * KDIM + 255) / 256), 256, 0, stream>>>(
        rgw_w, ugw_w, ogw_w, rgw_b, ugw_b, ogw_b, Wc, wcb);
    xconv_kernel<<<dim3(HW / 256, 192, BATCH), 256, 0, stream>>>(
        x, rg_w, rg_b, ug_w, ug_b, og_w, og_b, xg);
    sample_kernel<<<dim3(HW / 256, LVL, BATCH), 256, 0, stream>>>(
        flow, prev, M);
    gemm_gru_kernel<<<dim3(HW / 16, BATCH), 128, 0, stream>>>(
        M, Wc, wcb, xg, prev, out);
}
